// Mamba2VisionMixer_57793079935260
// MI455X (gfx1250) — matrix operand order repaired and checked by reading, not compiled, not run on hardware
//
#include <hip/hip_runtime.h>
#include <hip/hip_bf16.h>

typedef __bf16 bf16_t;
typedef __attribute__((ext_vector_type(16))) __bf16 v16bf;
typedef __attribute__((ext_vector_type(8)))  float  v8f;

// ---------- helpers ----------------------------------------------------------

__device__ __forceinline__ bf16_t f2bf(float f) {
    unsigned u = __builtin_bit_cast(unsigned, f);
    unsigned r = u + 0x7FFFu + ((u >> 16) & 1u);   // round-to-nearest-even
    unsigned short h = (unsigned short)(r >> 16);
    return __builtin_bit_cast(bf16_t, h);
}

__device__ __forceinline__ v8f wmma_bf16(v16bf a, v16bf b, v8f c) {
    return __builtin_amdgcn_wmma_f32_16x16x32_bf16(
        /*neg_a=*/false, a, /*neg_b=*/false, b,
        /*c_mod=*/(short)0, c, /*reuse_a=*/false, /*reuse_b=*/false);
}

// A fragment: 16x32 bf16, row-major M x K tile in LDS (ld = row stride in halves)
// lane<16: M=lane, K = 0..7 and 16..23 ; lane>=16: M=lane-16, K = 8..15 and 24..31
__device__ __forceinline__ v16bf frag_a(const bf16_t* p, int ld, int lane) {
    const bf16_t* r = p + (lane & 15) * ld + (((lane >> 4) & 1) << 3);
    v16bf f;
#pragma unroll
    for (int j = 0; j < 8; ++j) { f[j] = r[j]; f[8 + j] = r[16 + j]; }
    return f;
}

// B fragment: 32x16 bf16 (K x N) stored N-major in LDS: element(K=k,N=n) at p[n*ld + k]
// lane<16: N=lane, K = 0..7 and 16..23 ; lane>=16: N=lane-16, K = 8..15 and 24..31
__device__ __forceinline__ v16bf frag_b_t(const bf16_t* p, int ld, int lane) {
    const bf16_t* r = p + (lane & 15) * ld + (((lane >> 4) & 1) << 3);
    v16bf f;
#pragma unroll
    for (int j = 0; j < 8; ++j) { f[j] = r[j]; f[8 + j] = r[16 + j]; }
    return f;
}

// C/D 16x16 f32 store: VGPR r -> M = 8*(lane>>4)+r, N = lane&15
__device__ __forceinline__ void store_c(float* __restrict__ C, int ldc, int M, int N,
                                        int row0, int col0, const v8f& acc, int lane) {
    int n  = col0 + (lane & 15);
    int mb = row0 + ((lane >> 4) << 3);
    if (n < N) {
#pragma unroll
        for (int r = 0; r < 8; ++r) {
            int m = mb + r;
            if (m < M) C[(size_t)m * ldc + n] = acc[r];
        }
    }
}

// ---------- f32 -> bf16 pre-conversion (plain, for A operands) ---------------
__global__ __launch_bounds__(256) void cvt_bf16_kernel(const float* __restrict__ in,
                                                       bf16_t* __restrict__ out, int n4) {
    int i = blockIdx.x * 256 + threadIdx.x;
    if (i < n4) {
        float4 v = ((const float4*)in)[i];
        bf16_t o[4] = {f2bf(v.x), f2bf(v.y), f2bf(v.z), f2bf(v.w)};
        ((uint2*)out)[i] = *(const uint2*)o;
    }
}

// ---------- f32 [K,N] -> bf16 transposed [N,K] (for B/weight operands) -------
// LDS-tiled 32x32 transpose; coalesced loads and stores. K%32==0, N%32==0.
__global__ __launch_bounds__(256) void cvt_bf16_t_kernel(const float* __restrict__ in,
                                                         bf16_t* __restrict__ out,
                                                         int K, int N) {
    __shared__ float tile[32][33];
    int n0 = blockIdx.x * 32, k0 = blockIdx.y * 32;
    int tx = threadIdx.x & 31, ty = threadIdx.x >> 5;   // ty 0..7
#pragma unroll
    for (int i = 0; i < 32; i += 8)
        tile[ty + i][tx] = in[(size_t)(k0 + ty + i) * N + n0 + tx];
    __syncthreads();
#pragma unroll
    for (int i = 0; i < 32; i += 8)
        out[(size_t)(n0 + ty + i) * K + k0 + tx] = f2bf(tile[tx][ty + i]);
}

// ---------- bf16 WMMA GEMM, double-buffered LDS, BK=64 -----------------------
// C[M,N] = A[M,K] @ B[K,N]; A row-major [M,K], B pre-transposed N-major [N,K].
// M%128==0, K%64==0.
__global__ __launch_bounds__(256) void wmma_gemm_kernel(
    const bf16_t* __restrict__ A, const bf16_t* __restrict__ Bt, float* __restrict__ C,
    int M, int N, int K) {
    __shared__ __align__(16) bf16_t sA[2][128][72];    // 128 x 64, +8 pad (144B rows)
    __shared__ __align__(16) bf16_t sBt[2][128][72];   // N-major: [n][k], +8 pad

    const int tid  = threadIdx.x;
    const int lane = tid & 31;
    const int wid  = tid >> 5;
    const int wm   = wid >> 1;        // 0..3  (32-row strip)
    const int wn   = wid & 1;         // 0..1  (64-col strip)
    const int m0   = blockIdx.y * 128;
    const int n0   = blockIdx.x * 128;

    v8f acc[2][4];
#pragma unroll
    for (int i = 0; i < 2; ++i)
#pragma unroll
        for (int j = 0; j < 4; ++j) acc[i][j] = (v8f){0, 0, 0, 0, 0, 0, 0, 0};

    // Symmetric hoisted per-thread pointers; both operands advance by 64/stage.
    // OOB B rows clamp to N-1 (duplicate data only feeds discarded outputs).
    const bf16_t* aP[4];
    const bf16_t* bP[4];
    int rowE[4], k8E[4];
#pragma unroll
    for (int it = 0; it < 4; ++it) {
        int e    = it * 256 + tid;              // 1024 x 8-half groups per operand
        rowE[it] = e >> 3;                      // 0..127
        k8E[it]  = (e & 7) << 3;                // 0..56
        aP[it]   = A + (size_t)(m0 + rowE[it]) * K + k8E[it];
        int nr   = n0 + rowE[it];
        if (nr >= N) nr = N - 1;                // clamp (loop-invariant, branch-free)
        bP[it]   = Bt + (size_t)nr * K + k8E[it];
    }

    auto load_tiles = [&](int buf) {
        uint4 av[4], bv[4];
#pragma unroll
        for (int it = 0; it < 4; ++it) {        // gather: 8 b128 loads in flight
            av[it] = *(const uint4*)aP[it];
            bv[it] = *(const uint4*)bP[it];
            aP[it] += 64;
            bP[it] += 64;
        }
#pragma unroll
        for (int it = 0; it < 4; ++it) {
            *(uint4*)&sA[buf][rowE[it]][k8E[it]]  = av[it];
            *(uint4*)&sBt[buf][rowE[it]][k8E[it]] = bv[it];
        }
    };

    auto compute = [&](int buf) {
#pragma unroll
        for (int ks = 0; ks < 64; ks += 32) {
            v16bf a0 = frag_a(&sA[buf][wm * 32 +  0][ks], 72, lane);
            v16bf a1 = frag_a(&sA[buf][wm * 32 + 16][ks], 72, lane);
#pragma unroll
            for (int jn = 0; jn < 4; ++jn) {
                v16bf bf = frag_b_t(&sBt[buf][wn * 64 + jn * 16][ks], 72, lane);
                acc[0][jn] = wmma_bf16(a0, bf, acc[0][jn]);
                acc[1][jn] = wmma_bf16(a1, bf, acc[1][jn]);
            }
        }
    };

    const int nstages = K >> 6;
    load_tiles(0);
    __syncthreads();
    int buf = 0;
    for (int s = 1; s < nstages; ++s) {
        if (s + 1 < nstages) {                  // gfx1250 global_prefetch_b8, 1 ahead
            __builtin_prefetch((const void*)(aP[0] + 64), 0, 3);
            __builtin_prefetch((const void*)(bP[0] + 64), 0, 3);
        }
        load_tiles(buf ^ 1);                    // fill next stage
        compute(buf);                           // 16 WMMAs on current stage
        __syncthreads();
        buf ^= 1;
    }
    compute(buf);

#pragma unroll
    for (int im = 0; im < 2; ++im)
#pragma unroll
        for (int jn = 0; jn < 4; ++jn)
            store_c(C, N, M, N, m0 + wm * 32 + im * 16, n0 + wn * 64 + jn * 16,
                    acc[im][jn], lane);
}

// ---------- depthwise conv (k=4, SAME pad_left=1) + SiLU ---------------------
// Emits u in f32 (SSD math) and bf16 (x_dbl GEMM operand); z in f32.
__global__ __launch_bounds__(256) void conv_silu_kernel(
    const float* __restrict__ xz,
    const float* __restrict__ kx, const float* __restrict__ bx,
    const float* __restrict__ kz, const float* __restrict__ bz,
    float* __restrict__ u, bf16_t* __restrict__ u_bf, float* __restrict__ z) {
    int col = blockIdx.x * 256 + threadIdx.x;   // 0..4095
    int row = blockIdx.y;                       // 0..16383 (b*4096+l)
    int b = row >> 12, l = row & 4095;
    bool isz = col >= 2048;
    int ch = isz ? col - 2048 : col;
    const float* k    = isz ? kz : kx;
    const float* bias = isz ? bz : bx;
    float acc = bias[ch];
#pragma unroll
    for (int i = 0; i < 4; ++i) {
        int ll = l - 1 + i;
        if (ll >= 0 && ll < 4096)
            acc += xz[((size_t)(b * 4096 + ll)) * 4096 + col] * k[i * 2048 + ch];
    }
    float s = acc / (1.f + __expf(-acc));
    size_t idx = (size_t)row * 2048 + ch;
    if (isz) {
        z[idx] = s;
    } else {
        u[idx] = s;
        u_bf[idx] = f2bf(s);
    }
}

// ---------- SSD intra-chunk kernel: one block per (b, chunk, head) ----------
__global__ __launch_bounds__(128) void ssd_chunk_kernel(
    const float* __restrict__ xdbl, const float* __restrict__ u,
    const float* __restrict__ dt_bias, const float* __restrict__ A_log,
    float* __restrict__ y, float* __restrict__ states,
    float* __restrict__ acs_out, float* __restrict__ alast_out) {
    const int c = blockIdx.x, h = blockIdx.y, b = blockIdx.z;
    __shared__ bf16_t sC[64][136];     // C chunk, row-major l x n   (WMMA A)
    __shared__ bf16_t sB[64][136];     // B chunk, [s][n] => N-major for G's B (N=s,K=n)
    __shared__ bf16_t sBdT[128][72];   // [n][s] = B[s][n]*decay[s]  (states B, N=n,K=s)
    __shared__ bf16_t sXT[64][72];     // [p][s] = x*dt transposed   (states A & Y_diag B)
    __shared__ bf16_t sGL[64][72];     // (G .* L) bf16, l x s       (Y_diag A)
    __shared__ float  sAcs[64];
    __shared__ float  sDt[64];

    const int tid = threadIdx.x, lane = tid & 31, wid = tid >> 5;
    const size_t rowbase = (size_t)(b * 4096 + c * 64);
    const float Ah = -__expf(A_log[h]);

    if (tid < 64) {
        float raw = xdbl[(rowbase + tid) * 288 + h] + dt_bias[h];
        sDt[tid] = (raw > 20.f) ? raw : log1pf(__expf(raw));   // softplus
    }
    __syncthreads();
    if (tid == 0) {
        float s = 0.f;
        for (int l = 0; l < 64; ++l) { s += Ah * sDt[l]; sAcs[l] = s; }
    }
    __syncthreads();
    const float Alast = sAcs[63];
    if (tid < 64) acs_out[((size_t)((b * 64 + c) * 32 + h)) * 64 + tid] = sAcs[tid];
    if (tid == 0) alast_out[(b * 64 + c) * 32 + h] = Alast;

    // load C, B (and decay-scaled transposed B)
    for (int l = 0; l < 64; ++l) {
        const float* rp = xdbl + (rowbase + l) * 288;
        float cb = rp[160 + tid];
        float bb = rp[32 + tid];
        sC[l][tid] = f2bf(cb);
        sB[l][tid] = f2bf(bb);
        sBdT[tid][l] = f2bf(bb * __expf(Alast - sAcs[l]));
    }
    // x*dt transposed: sXT[p][s]
    for (int i = 0; i < 32; ++i) {
        int e = i * 128 + tid;
        int s = e >> 6, p = e & 63;
        sXT[p][s] = f2bf(u[(rowbase + s) * 2048 + h * 64 + p] * sDt[s]);
    }
    __syncthreads();

    const int l0 = wid * 16;
    // Phase 1: G = C @ B^T (64x64, K=128), apply segment decay mask -> sGL
#pragma unroll
    for (int jn = 0; jn < 4; ++jn) {
        v8f acc = (v8f){0, 0, 0, 0, 0, 0, 0, 0};
#pragma unroll
        for (int kk = 0; kk < 128; kk += 32)
            acc = wmma_bf16(frag_a(&sC[l0][kk], 136, lane),
                            frag_b_t(&sB[jn * 16][kk], 136, lane), acc);
        int scol = jn * 16 + (lane & 15);
        int lr0  = l0 + ((lane >> 4) << 3);
#pragma unroll
        for (int r = 0; r < 8; ++r) {
            int lr = lr0 + r;
            float v = (lr >= scol) ? acc[r] * __expf(sAcs[lr] - sAcs[scol]) : 0.f;
            sGL[lr][scol] = f2bf(v);
        }
    }
    __syncthreads();

    // Phase 2: Y_diag = (G.*L) @ X  (64x64, K=64) -> y
#pragma unroll
    for (int jp = 0; jp < 4; ++jp) {
        v8f acc = (v8f){0, 0, 0, 0, 0, 0, 0, 0};
#pragma unroll
        for (int kk = 0; kk < 64; kk += 32)
            acc = wmma_bf16(frag_a(&sGL[l0][kk], 72, lane),
                            frag_b_t(&sXT[jp * 16][kk], 72, lane), acc);
        int p   = jp * 16 + (lane & 15);
        int lr0 = l0 + ((lane >> 4) << 3);
#pragma unroll
        for (int r = 0; r < 8; ++r)
            y[(rowbase + lr0 + r) * 2048 + h * 64 + p] = acc[r];
    }

    // Phase 3: states = X^T @ (B .* decay)  (64p x 128n, K=64)
    const int p0 = wid * 16;
    const size_t sbase = ((size_t)((b * 64 + c) * 32 + h)) * 8192;
#pragma unroll
    for (int jn = 0; jn < 8; ++jn) {
        v8f acc = (v8f){0, 0, 0, 0, 0, 0, 0, 0};
#pragma unroll
        for (int kk = 0; kk < 64; kk += 32)
            acc = wmma_bf16(frag_a(&sXT[p0][kk], 72, lane),
                            frag_b_t(&sBdT[jn * 16][kk], 72, lane), acc);
        int n   = jn * 16 + (lane & 15);
        int pr0 = p0 + ((lane >> 4) << 3);
#pragma unroll
        for (int r = 0; r < 8; ++r)
            states[sbase + (size_t)(pr0 + r) * 128 + n] = acc[r];
    }
}

// ---------- inter-chunk state recurrence ------------------------------------
// prev[c] = prev[c-1]*exp(Alast[c-1]) + states[c-1]
__global__ __launch_bounds__(256) void chunk_scan_kernel(
    const float* __restrict__ states, const float* __restrict__ alast,
    float* __restrict__ prev) {
    int bh = blockIdx.x;             // 0..127
    int b = bh >> 5, h = bh & 31;
    int t = threadIdx.x;
    float carry[32];
#pragma unroll
    for (int i = 0; i < 32; ++i) carry[i] = 0.f;
    for (int c = 0; c < 64; ++c) {
        size_t base = ((size_t)((b * 64 + c) * 32 + h)) * 8192;
        float d = __expf(alast[(b * 64 + c) * 32 + h]);
#pragma unroll
        for (int i = 0; i < 32; ++i) {
            size_t idx = base + (size_t)i * 256 + t;
            prev[idx] = carry[i];
            carry[i] = carry[i] * d + states[idx];
        }
    }
}

// ---------- Y_off = (C @ prev^T) .* exp(A_cs) + skip u*D --------------------
__global__ __launch_bounds__(128) void yoff_kernel(
    const float* __restrict__ xdbl, const float* __restrict__ prev,
    const float* __restrict__ acs, const float* __restrict__ u,
    const float* __restrict__ D, float* __restrict__ y) {
    const int c = blockIdx.x, h = blockIdx.y, b = blockIdx.z;
    __shared__ bf16_t sC[64][136];
    __shared__ bf16_t sP[64][136];   // [p][n] natural => N-major for B (N=p, K=n)
    __shared__ float  sAcs[64];
    const int tid = threadIdx.x, lane = tid & 31, wid = tid >> 5;
    const size_t rowbase = (size_t)(b * 4096 + c * 64);
    const size_t sbase   = ((size_t)((b * 64 + c) * 32 + h)) * 8192;

    if (tid < 64) sAcs[tid] = acs[((size_t)((b * 64 + c) * 32 + h)) * 64 + tid];
    for (int l = 0; l < 64; ++l)
        sC[l][tid] = f2bf(xdbl[(rowbase + l) * 288 + 160 + tid]);
    for (int p = 0; p < 64; ++p)
        sP[p][tid] = f2bf(prev[sbase + (size_t)p * 128 + tid]);
    __syncthreads();

    const int l0 = wid * 16;
    const float Dh = D[h];
#pragma unroll
    for (int jp = 0; jp < 4; ++jp) {
        v8f acc = (v8f){0, 0, 0, 0, 0, 0, 0, 0};
#pragma unroll
        for (int kk = 0; kk < 128; kk += 32)
            acc = wmma_bf16(frag_a(&sC[l0][kk], 136, lane),
                            frag_b_t(&sP[jp * 16][kk], 136, lane), acc);
        int p   = jp * 16 + (lane & 15);
        int lr0 = l0 + ((lane >> 4) << 3);
#pragma unroll
        for (int r = 0; r < 8; ++r) {
            int lr = lr0 + r;
            size_t idx = (rowbase + lr) * 2048 + h * 64 + p;
            y[idx] = y[idx] + acc[r] * __expf(sAcs[lr]) + u[idx] * Dh;
        }
    }
}

// ---------- gating (y * silu(z)) + RMS norm -> bf16 (GEMM operand) ----------
__global__ __launch_bounds__(256) void gate_norm_kernel(
    const float* __restrict__ y, const float* __restrict__ z,
    const float* __restrict__ norm_w, bf16_t* __restrict__ yn) {
    int row = blockIdx.x;
    int tid = threadIdx.x;
    __shared__ float red[256];
    float g[8];
    float ss = 0.f;
#pragma unroll
    for (int i = 0; i < 8; ++i) {
        int ch = tid + i * 256;
        float zz = z[(size_t)row * 2048 + ch];
        float gv = y[(size_t)row * 2048 + ch] * (zz / (1.f + __expf(-zz)));
        g[i] = gv; ss += gv * gv;
    }
    red[tid] = ss; __syncthreads();
    for (int off = 128; off > 0; off >>= 1) {
        if (tid < off) red[tid] += red[tid + off];
        __syncthreads();
    }
    float scale = rsqrtf(red[0] / 2048.f + 1e-6f);
#pragma unroll
    for (int i = 0; i < 8; ++i) {
        int ch = tid + i * 256;
        yn[(size_t)row * 2048 + ch] = f2bf(g[i] * scale * norm_w[ch]);
    }
}

// ---------- launcher ---------------------------------------------------------

extern "C" void kernel_launch(void* const* d_in, const int* in_sizes, int n_in,
                              void* d_out, int out_size, void* d_ws, size_t ws_size,
                              hipStream_t stream) {
    (void)in_sizes; (void)n_in; (void)out_size; (void)ws_size;
    const float* x        = (const float*)d_in[0];
    const float* W_in     = (const float*)d_in[1];
    const float* conv_x_k = (const float*)d_in[2];
    const float* conv_x_b = (const float*)d_in[3];
    const float* conv_z_k = (const float*)d_in[4];
    const float* conv_z_b = (const float*)d_in[5];
    const float* W_x      = (const float*)d_in[6];
    const float* dt_bias  = (const float*)d_in[7];
    const float* A_log    = (const float*)d_in[8];
    const float* Dp       = (const float*)d_in[9];
    const float* norm_w   = (const float*)d_in[10];
    const float* W_out    = (const float*)d_in[11];
    float* out = (float*)d_out;

    float* ws = (float*)d_ws;
    // f32 workspace layout (floats). states reuses the xz region (xz dead after conv).
    const size_t OFF_XZ     = 0;                               // 16384*4096
    const size_t OFF_U      = 67108864;                        // 16384*2048
    const size_t OFF_Z      = OFF_U + 33554432;
    const size_t OFF_XDBL   = OFF_Z + 33554432;                // 16384*288
    const size_t OFF_Y      = OFF_XDBL + 4718592;              // 16384*2048
    const size_t OFF_STATES = OFF_XZ;                          // 4*64*32*8192
    const size_t OFF_PREV   = OFF_Y + 33554432;                // 4*64*32*8192
    const size_t OFF_ACS    = OFF_PREV + 67108864;             // 4*64*32*64
    const size_t OFF_ALAST  = OFF_ACS + 524288;                // 4*64*32
    const size_t OFF_BF     = OFF_ALAST + 8192;                // bf16 region below

    float* xz     = ws + OFF_XZ;
    float* u      = ws + OFF_U;
    float* zb     = ws + OFF_Z;
    float* xdbl   = ws + OFF_XDBL;
    float* y      = ws + OFF_Y;
    float* states = ws + OFF_STATES;
    float* prev   = ws + OFF_PREV;
    float* acs    = ws + OFF_ACS;
    float* alast  = ws + OFF_ALAST;

    // bf16 workspace (halves). Weights are stored TRANSPOSED (N-major [N,K]).
    bf16_t* hb      = (bf16_t*)(ws + OFF_BF);
    bf16_t* x_bf    = hb;                    // 16384*1024        (A, row-major)
    bf16_t* Win_bf  = x_bf   + 16777216;     // [4096][1024]      (B, transposed)
    bf16_t* Wx_bf   = Win_bf + 4194304;      // [288][2048]       (B, transposed)
    bf16_t* Wout_bf = Wx_bf  + 589824;       // [1024][2048]      (B, transposed)
    bf16_t* u_bf    = Wout_bf + 2097152;     // 16384*2048        (A, row-major)
    bf16_t* yn_bf   = u_bf   + 33554432;     // 16384*2048        (A, row-major)

    // 0) pre-convert: A operands plain bf16; weights bf16 + transpose to N-major
    cvt_bf16_kernel<<<16384, 256, 0, stream>>>(x, x_bf, 4194304);
    cvt_bf16_t_kernel<<<dim3(4096 / 32, 1024 / 32), 256, 0, stream>>>(
        W_in, Win_bf, 1024, 4096);
    cvt_bf16_t_kernel<<<dim3(288 / 32, 2048 / 32), 256, 0, stream>>>(
        W_x, Wx_bf, 2048, 288);
    cvt_bf16_t_kernel<<<dim3(1024 / 32, 2048 / 32), 256, 0, stream>>>(
        W_out, Wout_bf, 2048, 1024);

    // 1) xz = x @ W_in   (16384x1024 @ 1024x4096)
    wmma_gemm_kernel<<<dim3(4096 / 128, 16384 / 128), 256, 0, stream>>>(
        x_bf, Win_bf, xz, 16384, 4096, 1024);
    // 2) depthwise conv + SiLU -> u (f32 + bf16), z
    conv_silu_kernel<<<dim3(16, 16384), 256, 0, stream>>>(
        xz, conv_x_k, conv_x_b, conv_z_k, conv_z_b, u, u_bf, zb);
    // 3) x_dbl = u @ W_x (16384x2048 @ 2048x288)
    wmma_gemm_kernel<<<dim3(3, 16384 / 128), 256, 0, stream>>>(
        u_bf, Wx_bf, xdbl, 16384, 288, 2048);
    // 4) SSD intra-chunk: Y_diag + per-chunk states
    ssd_chunk_kernel<<<dim3(64, 32, 4), 128, 0, stream>>>(
        xdbl, u, dt_bias, A_log, y, states, acs, alast);
    // 5) inter-chunk recurrence -> prev_states
    chunk_scan_kernel<<<128, 256, 0, stream>>>(states, alast, prev);
    // 6) Y_off + skip u*D
    yoff_kernel<<<dim3(64, 32, 4), 128, 0, stream>>>(xdbl, prev, acs, u, Dp, y);
    // 7) gating + RMS norm -> bf16
    gate_norm_kernel<<<16384, 256, 0, stream>>>(y, zb, norm_w, yn_bf);
    // 8) out = yn @ W_out (16384x2048 @ 2048x1024)
    wmma_gemm_kernel<<<dim3(1024 / 128, 16384 / 128), 256, 0, stream>>>(
        yn_bf, Wout_bf, out, 16384, 1024, 2048);
}
